// GroupedLinear_11269994184722
// MI455X (gfx1250) — matrix operand order repaired and checked by reading, not compiled, not run on hardware
//
#include <hip/hip_runtime.h>
#include <stdint.h>

typedef __bf16 bf16_t;
typedef bf16_t v16bf __attribute__((ext_vector_type(16)));
typedef float  v8f   __attribute__((ext_vector_type(8)));
typedef int    v4i   __attribute__((ext_vector_type(4)));

#if __has_builtin(__builtin_amdgcn_global_load_async_to_lds_b128) && \
    __has_builtin(__builtin_amdgcn_s_wait_asynccnt)
#define USE_ASYNC 1
#else
#define USE_ASYNC 0
#endif

#define AS1 __attribute__((address_space(1)))
#define AS3 __attribute__((address_space(3)))

namespace {
constexpr int IN_F    = 1024;
constexpr int OUT_F   = 1024;
constexpr int BM      = 128;
constexpr int BN      = 64;
constexpr int BK      = 32;          // one WMMA K-step per stage
constexpr int APAD    = BK + 8;      // 40 bf16 -> 80B row stride
constexpr int BPAD    = BN + 8;      // 72 bf16 -> 144B row stride
constexpr int NK      = IN_F / BK;   // 32
constexpr int THREADS = 256;         // 8 waves (wave32)
}

// ---------- fp32 -> bf16 hi/lo split, pure 32-bit integer ops ----------
__device__ __forceinline__ unsigned int bf16_rne_bits(unsigned int u) {
  return u + 0x7FFFu + ((u >> 16) & 1u);
}

struct Bf16Pair4 { uint2 hi, lo; };

__device__ __forceinline__ Bf16Pair4 split4(float4 v) {
  unsigned int u0 = __float_as_uint(v.x), u1 = __float_as_uint(v.y);
  unsigned int u2 = __float_as_uint(v.z), u3 = __float_as_uint(v.w);
  unsigned int r0 = bf16_rne_bits(u0), r1 = bf16_rne_bits(u1);
  unsigned int r2 = bf16_rne_bits(u2), r3 = bf16_rne_bits(u3);
  unsigned int t0 = __float_as_uint(v.x - __uint_as_float(r0 & 0xFFFF0000u));
  unsigned int t1 = __float_as_uint(v.y - __uint_as_float(r1 & 0xFFFF0000u));
  unsigned int t2 = __float_as_uint(v.z - __uint_as_float(r2 & 0xFFFF0000u));
  unsigned int t3 = __float_as_uint(v.w - __uint_as_float(r3 & 0xFFFF0000u));
  unsigned int s0 = bf16_rne_bits(t0), s1 = bf16_rne_bits(t1);
  unsigned int s2 = bf16_rne_bits(t2), s3 = bf16_rne_bits(t3);
  Bf16Pair4 p;
  p.hi.x = __builtin_amdgcn_perm(r1, r0, 0x07060302u);
  p.hi.y = __builtin_amdgcn_perm(r3, r2, 0x07060302u);
  p.lo.x = __builtin_amdgcn_perm(s1, s0, 0x07060302u);
  p.lo.y = __builtin_amdgcn_perm(s3, s2, 0x07060302u);
  return p;
}

union FragU { v16bf v; uint4 q[2]; };

__device__ __forceinline__ void async_b128(const unsigned short* g, unsigned short* l) {
#if USE_ASYNC
  __builtin_amdgcn_global_load_async_to_lds_b128(
      (AS1 v4i*)(uintptr_t)g, (AS3 v4i*)l, 0, 0);
#else
  (void)g; (void)l;
#endif
}

// =====================================================================================
// Phase 1: one-shot fp32 -> packed bf16 hi/lo planes
// =====================================================================================
__global__ __launch_bounds__(256)
void cvt_split_kernel(const float* __restrict__ src,
                      uint2* __restrict__ hiDst, uint2* __restrict__ loDst,
                      int nVec4)
{
  int i = blockIdx.x * blockDim.x + threadIdx.x;
  if (i >= nVec4) return;
  float4 v = ((const float4*)src)[i];
  Bf16Pair4 p = split4(v);
  hiDst[i] = p.hi;
  loDst[i] = p.lo;
}

// =====================================================================================
// Phase 2: grouped GEMM on pre-split bf16 planes (bf16x3 error-compensated WMMA)
// =====================================================================================
__global__ __launch_bounds__(THREADS)
void grouped_gemm_presplit(const unsigned short* __restrict__ Xh,
                           const unsigned short* __restrict__ Xl,
                           const unsigned short* __restrict__ Wh,
                           const unsigned short* __restrict__ Wl,
                           const long long* __restrict__ offs,
                           float* __restrict__ Y,
                           int tokens, int groups)
{
  __shared__ __align__(16) unsigned short sAh[2][BM * APAD];
  __shared__ __align__(16) unsigned short sAl[2][BM * APAD];
  __shared__ __align__(16) unsigned short sBh[2][BK * BPAD];
  __shared__ __align__(16) unsigned short sBl[2][BK * BPAD];

  const int tid  = threadIdx.x;
  const int lane = tid & 31;
  const int wave = tid >> 5;
  const int wm   = wave & 3;                 // 4 waves along M (32 rows each)
  const int wn   = wave >> 2;                // 2 waves along N (32 cols each)
  const int rowBase = blockIdx.y * BM;
  const int colBase = blockIdx.x * BN;

  // A tile: 128x32 bf16 per plane = 512 uint4; 2 per thread (8 bf16 along K each)
  int aRow[2], aC[2];
#pragma unroll
  for (int i = 0; i < 2; ++i) {
    int idx = tid + i * THREADS;
    aRow[i] = idx >> 2;
    aC[i]   = (idx & 3) << 3;
  }
  // B tile: 32x64 bf16 per plane = 256 uint4; 1 per thread
  const int bRow = tid >> 3;
  const int bC   = (tid & 7) << 3;

  auto group_of = [&](int r) -> int {
    int g = 0;
    for (int j = 0; j < groups; ++j) g += (offs[j] <= (long long)r) ? 1 : 0;
    return g;
  };
  const int gLo = group_of(rowBase);
  const int gHi = group_of(rowBase + BM - 1);

  v8f acc[2][2];
#pragma unroll
  for (int mt = 0; mt < 2; ++mt)
#pragma unroll
    for (int nt = 0; nt < 2; ++nt)
#pragma unroll
      for (int e = 0; e < 8; ++e) acc[mt][nt][e] = 0.0f;

  // ---- shared compute step: DS fragment loads + 12 WMMA on stage st ----
  auto compute_stage = [&](int st) {
    const int m  = lane & 15;
    const int hk = (lane >> 4) << 3;         // 0 or 8
    FragU ah[2], al[2];
#pragma unroll
    for (int mt = 0; mt < 2; ++mt) {
      const int r = wm * 32 + mt * 16 + m;
      const unsigned short* ph = &sAh[st][r * APAD];
      const unsigned short* pl = &sAl[st][r * APAD];
      ah[mt].q[0] = *(const uint4*)(ph + hk);
      ah[mt].q[1] = *(const uint4*)(ph + 16 + hk);
      al[mt].q[0] = *(const uint4*)(pl + hk);
      al[mt].q[1] = *(const uint4*)(pl + 16 + hk);
    }
#pragma unroll
    for (int nt = 0; nt < 2; ++nt) {
      FragU bh, bl;
      const unsigned short* ph = &sBh[st][wn * 32 + nt * 16 + m];
      const unsigned short* pl = &sBl[st][wn * 32 + nt * 16 + m];
#pragma unroll
      for (int s = 0; s < 16; ++s) {
        const int k = (s < 8) ? (hk + s) : (16 + hk + (s - 8));
        ((unsigned short*)&bh)[s] = ph[k * BPAD];
        ((unsigned short*)&bl)[s] = pl[k * BPAD];
      }
#pragma unroll
      for (int mt = 0; mt < 2; ++mt) {
        acc[mt][nt] = __builtin_amdgcn_wmma_f32_16x16x32_bf16(
            false, ah[mt].v, false, bh.v, (short)0, acc[mt][nt], false, false);
        acc[mt][nt] = __builtin_amdgcn_wmma_f32_16x16x32_bf16(
            false, al[mt].v, false, bh.v, (short)0, acc[mt][nt], false, false);
        acc[mt][nt] = __builtin_amdgcn_wmma_f32_16x16x32_bf16(
            false, ah[mt].v, false, bl.v, (short)0, acc[mt][nt], false, false);
      }
    }
  };

#if USE_ASYNC
  if (gLo == gHi) {
    // ---------- FAST PATH: whole block in one group -> async global->LDS fills ----------
    const int g = gLo;
    const size_t wOff = (size_t)g * IN_F * OUT_F;

    auto issue_stage = [&](int s, int k0) {
#pragma unroll
      for (int i = 0; i < 2; ++i) {
        size_t o = (size_t)(rowBase + aRow[i]) * IN_F + k0 + aC[i];
        async_b128(Xh + o, &sAh[s][aRow[i] * APAD + aC[i]]);
        async_b128(Xl + o, &sAl[s][aRow[i] * APAD + aC[i]]);
      }
      size_t o = wOff + (size_t)(k0 + bRow) * OUT_F + colBase + bC;
      async_b128(Wh + o, &sBh[s][bRow * BPAD + bC]);
      async_b128(Wl + o, &sBl[s][bRow * BPAD + bC]);
    };

    issue_stage(0, 0);
    __builtin_amdgcn_s_wait_asynccnt(0);
    __syncthreads();

    int st = 0;
    for (int kt = 0; kt < NK; ++kt) {
      const bool more = (kt + 1 < NK);
      if (more) issue_stage(st ^ 1, (kt + 1) * BK);  // overlap copy with compute
      compute_stage(st);
      if (more) __builtin_amdgcn_s_wait_asynccnt(0);
      __syncthreads();
      st ^= 1;
    }
  } else
#endif
  {
    // ---------- MASKED PATH: block straddles a group boundary ----------
    for (int g = gLo; g <= gHi; ++g) {
      unsigned int amask[2];
#pragma unroll
      for (int i = 0; i < 2; ++i)
        amask[i] = (group_of(rowBase + aRow[i]) == g) ? 0xFFFFFFFFu : 0u;

      const size_t wOff = (size_t)g * IN_F * OUT_F;

      uint4 aRh[2], aRl[2], bRh, bRl;
      auto load_regs = [&](int k0) {
#pragma unroll
        for (int i = 0; i < 2; ++i) {
          size_t o = (size_t)(rowBase + aRow[i]) * IN_F + k0 + aC[i];
          aRh[i] = *(const uint4*)(Xh + o);
          aRl[i] = *(const uint4*)(Xl + o);
        }
        size_t o = wOff + (size_t)(k0 + bRow) * OUT_F + colBase + bC;
        bRh = *(const uint4*)(Wh + o);
        bRl = *(const uint4*)(Wl + o);
      };
      auto store_stage = [&](int s) {
#pragma unroll
        for (int i = 0; i < 2; ++i) {
          uint4 h = aRh[i], l = aRl[i];
          h.x &= amask[i]; h.y &= amask[i]; h.z &= amask[i]; h.w &= amask[i];
          l.x &= amask[i]; l.y &= amask[i]; l.z &= amask[i]; l.w &= amask[i];
          *(uint4*)(&sAh[s][aRow[i] * APAD + aC[i]]) = h;
          *(uint4*)(&sAl[s][aRow[i] * APAD + aC[i]]) = l;
        }
        *(uint4*)(&sBh[s][bRow * BPAD + bC]) = bRh;
        *(uint4*)(&sBl[s][bRow * BPAD + bC]) = bRl;
      };

      load_regs(0);
      store_stage(0);
      __syncthreads();

      int st = 0;
      for (int kt = 0; kt < NK; ++kt) {
        const bool more = (kt + 1 < NK);
        if (more) load_regs((kt + 1) * BK);
        compute_stage(st);
        if (more) store_stage(st ^ 1);
        __syncthreads();
        st ^= 1;
      }
    }
  }

  // Epilogue: VGPR v: lanes 0-15 -> M=v, lanes 16-31 -> M=v+8; N=lane&15
  const int nLoc = lane & 15;
  const int mOff = (lane < 16) ? 0 : 8;
#pragma unroll
  for (int mt = 0; mt < 2; ++mt)
#pragma unroll
    for (int nt = 0; nt < 2; ++nt)
#pragma unroll
      for (int v = 0; v < 8; ++v) {
        int row = rowBase + wm * 32 + mt * 16 + mOff + v;
        int col = colBase + wn * 32 + nt * 16 + nLoc;
        Y[(size_t)row * OUT_F + col] = acc[mt][nt][v];
      }
}

// =====================================================================================
// Fallback: fused conversion GEMM (used only if d_ws too small)
// =====================================================================================
__global__ __launch_bounds__(THREADS)
void grouped_gemm_fused(const float* __restrict__ X,
                        const float* __restrict__ W,
                        const long long* __restrict__ offs,
                        float* __restrict__ Y,
                        int tokens, int groups)
{
  __shared__ __align__(16) unsigned short sAh[2][BM * APAD];
  __shared__ __align__(16) unsigned short sAl[2][BM * APAD];
  __shared__ __align__(16) unsigned short sBh[2][BK * BPAD];
  __shared__ __align__(16) unsigned short sBl[2][BK * BPAD];

  const int tid  = threadIdx.x;
  const int lane = tid & 31;
  const int wave = tid >> 5;
  const int wm   = wave & 3;
  const int wn   = wave >> 2;
  const int rowBase = blockIdx.y * BM;
  const int colBase = blockIdx.x * BN;

  int aRow[4], aC[4];
#pragma unroll
  for (int i = 0; i < 4; ++i) {
    int idx = tid + i * THREADS;
    aRow[i] = idx >> 3; aC[i] = (idx & 7) << 2;
  }
  int bRow[2], bC[2];
#pragma unroll
  for (int i = 0; i < 2; ++i) {
    int idx = tid + i * THREADS;
    bRow[i] = idx >> 4; bC[i] = (idx & 15) << 2;
  }

  auto group_of = [&](int r) -> int {
    int g = 0;
    for (int j = 0; j < groups; ++j) g += (offs[j] <= (long long)r) ? 1 : 0;
    return g;
  };
  const int gLo = group_of(rowBase);
  const int gHi = group_of(rowBase + BM - 1);

  v8f acc[2][2];
#pragma unroll
  for (int mt = 0; mt < 2; ++mt)
#pragma unroll
    for (int nt = 0; nt < 2; ++nt)
#pragma unroll
      for (int e = 0; e < 8; ++e) acc[mt][nt][e] = 0.0f;

  for (int g = gLo; g <= gHi; ++g) {
    unsigned int amask[4];
#pragma unroll
    for (int i = 0; i < 4; ++i)
      amask[i] = (group_of(rowBase + aRow[i]) == g) ? 0xFFFFFFFFu : 0u;

    const float* __restrict__ Wg = W + (size_t)g * IN_F * OUT_F;

    float4 aR[4]; float4 bR[2];
    auto load_regs = [&](int k0) {
#pragma unroll
      for (int i = 0; i < 4; ++i)
        aR[i] = *(const float4*)(X + (size_t)(rowBase + aRow[i]) * IN_F + k0 + aC[i]);
#pragma unroll
      for (int i = 0; i < 2; ++i)
        bR[i] = *(const float4*)(Wg + (size_t)(k0 + bRow[i]) * OUT_F + colBase + bC[i]);
    };
    auto store_stage = [&](int s) {
#pragma unroll
      for (int i = 0; i < 4; ++i) {
        Bf16Pair4 p = split4(aR[i]);
        p.hi.x &= amask[i]; p.hi.y &= amask[i];
        p.lo.x &= amask[i]; p.lo.y &= amask[i];
        *(uint2*)(&sAh[s][aRow[i] * APAD + aC[i]]) = p.hi;
        *(uint2*)(&sAl[s][aRow[i] * APAD + aC[i]]) = p.lo;
      }
#pragma unroll
      for (int i = 0; i < 2; ++i) {
        Bf16Pair4 p = split4(bR[i]);
        *(uint2*)(&sBh[s][bRow[i] * BPAD + bC[i]]) = p.hi;
        *(uint2*)(&sBl[s][bRow[i] * BPAD + bC[i]]) = p.lo;
      }
    };

    load_regs(0);
    store_stage(0);
    __syncthreads();

    int st = 0;
    for (int kt = 0; kt < NK; ++kt) {
      const bool more = (kt + 1 < NK);
      if (more) load_regs((kt + 1) * BK);

      const int m  = lane & 15;
      const int hk = (lane >> 4) << 3;
      FragU ah[2], al[2];
#pragma unroll
      for (int mt = 0; mt < 2; ++mt) {
        const int r = wm * 32 + mt * 16 + m;
        const unsigned short* ph = &sAh[st][r * APAD];
        const unsigned short* pl = &sAl[st][r * APAD];
        ah[mt].q[0] = *(const uint4*)(ph + hk);
        ah[mt].q[1] = *(const uint4*)(ph + 16 + hk);
        al[mt].q[0] = *(const uint4*)(pl + hk);
        al[mt].q[1] = *(const uint4*)(pl + 16 + hk);
      }
#pragma unroll
      for (int nt = 0; nt < 2; ++nt) {
        FragU bh, bl;
        const unsigned short* ph = &sBh[st][wn * 32 + nt * 16 + m];
        const unsigned short* pl = &sBl[st][wn * 32 + nt * 16 + m];
#pragma unroll
        for (int s = 0; s < 16; ++s) {
          const int k = (s < 8) ? (hk + s) : (16 + hk + (s - 8));
          ((unsigned short*)&bh)[s] = ph[k * BPAD];
          ((unsigned short*)&bl)[s] = pl[k * BPAD];
        }
#pragma unroll
        for (int mt = 0; mt < 2; ++mt) {
          acc[mt][nt] = __builtin_amdgcn_wmma_f32_16x16x32_bf16(
              false, ah[mt].v, false, bh.v, (short)0, acc[mt][nt], false, false);
          acc[mt][nt] = __builtin_amdgcn_wmma_f32_16x16x32_bf16(
              false, al[mt].v, false, bh.v, (short)0, acc[mt][nt], false, false);
          acc[mt][nt] = __builtin_amdgcn_wmma_f32_16x16x32_bf16(
              false, ah[mt].v, false, bl.v, (short)0, acc[mt][nt], false, false);
        }
      }
      if (more) store_stage(st ^ 1);
      __syncthreads();
      st ^= 1;
    }
  }

  const int nLoc = lane & 15;
  const int mOff = (lane < 16) ? 0 : 8;
#pragma unroll
  for (int mt = 0; mt < 2; ++mt)
#pragma unroll
    for (int nt = 0; nt < 2; ++nt)
#pragma unroll
      for (int v = 0; v < 8; ++v) {
        int row = rowBase + wm * 32 + mt * 16 + mOff + v;
        int col = colBase + wn * 32 + nt * 16 + nLoc;
        Y[(size_t)row * OUT_F + col] = acc[mt][nt][v];
      }
}

// =====================================================================================
extern "C" void kernel_launch(void* const* d_in, const int* in_sizes, int n_in,
                              void* d_out, int out_size, void* d_ws, size_t ws_size,
                              hipStream_t stream) {
  const float*     X    = (const float*)d_in[0];
  const float*     W    = (const float*)d_in[1];
  const long long* offs = (const long long*)d_in[2];
  float*           Y    = (float*)d_out;
  const int groups = in_sizes[2];
  const int tokens = in_sizes[0] / IN_F;        // 16384

  const size_t xElems = (size_t)tokens * IN_F;
  const size_t wElems = (size_t)groups * IN_F * OUT_F;
  const size_t need   = (xElems + wElems) * 2 * sizeof(unsigned short);

  dim3 grid(OUT_F / BN, tokens / BM);           // (16, 128)

  if (ws_size >= need) {
    unsigned short* Xh = (unsigned short*)d_ws;
    unsigned short* Xl = Xh + xElems;
    unsigned short* Wh = Xl + xElems;
    unsigned short* Wl = Wh + wElems;

    const int xV4 = (int)(xElems / 4);
    const int wV4 = (int)(wElems / 4);
    hipLaunchKernelGGL(cvt_split_kernel, dim3((xV4 + 255) / 256), dim3(256), 0, stream,
                       X, (uint2*)Xh, (uint2*)Xl, xV4);
    hipLaunchKernelGGL(cvt_split_kernel, dim3((wV4 + 255) / 256), dim3(256), 0, stream,
                       W, (uint2*)Wh, (uint2*)Wl, wV4);
    hipLaunchKernelGGL(grouped_gemm_presplit, grid, dim3(THREADS), 0, stream,
                       Xh, Xl, Wh, Wl, offs, Y, tokens, groups);
  } else {
    hipLaunchKernelGGL(grouped_gemm_fused, grid, dim3(THREADS), 0, stream,
                       X, W, offs, Y, tokens, groups);
  }
}
